// ConflictModel_64089501991217
// MI455X (gfx1250) — matrix operand order repaired and checked by reading, not compiled, not run on hardware
//
#include <hip/hip_runtime.h>
#include <hip/hip_bf16.h>
#include <math.h>

// ---------------------------------------------------------------------------
// ConflictModel on gfx1250 (MI455X): bf16 WMMA pipeline, one wave32 per batch
// element. B=4096, A=16 agents, N=128 cities, E=128, H=8 heads, dh=16.
// All LDS is wave-private; CDNA5 LDS ops are in-order per wave, so only
// compiler-level fences (wave_barrier) are needed, no s_barrier.
// ---------------------------------------------------------------------------

typedef __bf16 bf16_t;
typedef __attribute__((ext_vector_type(16))) __bf16 v16bf;
typedef __attribute__((ext_vector_type(8)))  __bf16 v8bf;
typedef __attribute__((ext_vector_type(4)))  __bf16 v4bf;
typedef __attribute__((ext_vector_type(8)))  float  v8f;

#define NEGV  (-1e9f)
#define CLIPV (10.0f)
#define WAVES 4
// per-wave LDS: 6 x (16x128 bf16 = 4096B) + 1024B scratch + 64B acts, padded
#define WAVE_BYTES 25728

static __device__ __forceinline__ void wavefence() {
  __builtin_amdgcn_wave_barrier();  // ordering fence; LDS is in-order per wave
}

static __device__ __forceinline__ v8f zero8f() {
  v8f z;
#pragma unroll
  for (int i = 0; i < 8; ++i) z[i] = 0.0f;
  return z;
}
static __device__ __forceinline__ v8bf zero8bf() {
  v8bf z;
#pragma unroll
  for (int i = 0; i < 8; ++i) z[i] = (bf16_t)0.0f;
  return z;
}
static __device__ __forceinline__ v16bf cat16(v8bf lo, v8bf hi) {
  return __builtin_shufflevector(lo, hi, 0, 1, 2, 3, 4, 5, 6, 7,
                                 8, 9, 10, 11, 12, 13, 14, 15);
}
static __device__ __forceinline__ v8f wmma_bf16(v16bf a, v16bf b, v8f c) {
  // D = A(16x32) * B(32x16) + C, f32 accumulate
  return __builtin_amdgcn_wmma_f32_16x16x32_bf16(false, a, false, b,
                                                 (short)0, c, false, false);
}

// ---- operand fragment builders (contiguous k = 16*hi + s, same in A and B) --------------

// A fragment: X is 16x128 row-major bf16 (LDS); 16x32 slice at K=32*kt.
static __device__ __forceinline__ v16bf lda128(const bf16_t* X, int kt, int lane) {
  const int M = lane & 15, hi = lane >> 4;
  const bf16_t* p = X + M * 128 + kt * 32 + hi * 16;
  return cat16(*(const v8bf*)p, *(const v8bf*)(p + 8));
}
// A fragment with real K=16 (zero-padded to 32): rows 16 x rowstride.
static __device__ __forceinline__ v16bf lda_k16(const bf16_t* X, int rowstride,
                                                int col0, int lane) {
  const int M = lane & 15, hi = lane >> 4;
  if (hi) return cat16(zero8bf(), zero8bf());
  const bf16_t* p = X + M * rowstride + col0;
  return cat16(*(const v8bf*)p, *(const v8bf*)(p + 8));
}
// B fragment from transposed weight WT[N=128][K=128] row-major (global/L2).
static __device__ __forceinline__ v16bf ldb_w(const bf16_t* WT, int nt, int kt,
                                              int lane) {
  const int N = lane & 15, hi = lane >> 4;
  const bf16_t* p = WT + (nt * 16 + N) * 128 + kt * 32 + hi * 16;
  return cat16(*(const v8bf*)p, *(const v8bf*)(p + 8));
}
// B = R^T with real K=16 (padded): lane column N reads row (rowbase+N) of R.
static __device__ __forceinline__ v16bf ldb_rT_k16(const bf16_t* R, int rowstride,
                                                   int rowbase, int col0, int lane) {
  const int N = lane & 15, hi = lane >> 4;
  if (hi) return cat16(zero8bf(), zero8bf());  // K=16..31 padded with zeros
  const bf16_t* p = R + (rowbase + N) * rowstride + col0;
  return cat16(*(const v8bf*)p, *(const v8bf*)(p + 8));
}
// B = R^T, full K=128 stepped by kt; R is 16x128 row-major (LDS).
static __device__ __forceinline__ v16bf ldb_rT(const bf16_t* R, int kt, int lane) {
  const int N = lane & 15, hi = lane >> 4;
  const bf16_t* p = R + N * 128 + kt * 32 + hi * 16;
  return cat16(*(const v8bf*)p, *(const v8bf*)(p + 8));
}

// ---- row reductions across the 16-lane halves (C-frag N dimension) ---------
static __device__ __forceinline__ float rsum16(float x) {
#pragma unroll
  for (int off = 1; off < 16; off <<= 1) x += __shfl_xor(x, off, 16);
  return x;
}
static __device__ __forceinline__ float rmax16(float x) {
#pragma unroll
  for (int off = 1; off < 16; off <<= 1) x = fmaxf(x, __shfl_xor(x, off, 16));
  return x;
}

// ---- 16x128 @ 128x128 matmul stages (forceinline => addrspace-specialized) -

static __device__ __forceinline__ void mm_store(const bf16_t* A, const bf16_t* WT,
                                                bf16_t* D, int lane) {
  v16bf af[4];
#pragma unroll
  for (int kt = 0; kt < 4; ++kt) af[kt] = lda128(A, kt, lane);
  const int N = lane & 15, hi = lane >> 4;
#pragma unroll
  for (int nt = 0; nt < 8; ++nt) {
    v8f acc = zero8f();
#pragma unroll
    for (int kt = 0; kt < 4; ++kt)
      acc = wmma_bf16(af[kt], ldb_w(WT, nt, kt, lane), acc);
#pragma unroll
    for (int j = 0; j < 8; ++j)
      D[(j + 8 * hi) * 128 + nt * 16 + N] = (bf16_t)acc[j];
  }
}

// store result transposed: DT is 128x16 (for v^T, feeding attn@v B-operand)
static __device__ __forceinline__ void mm_storeT(const bf16_t* A, const bf16_t* WT,
                                                 bf16_t* DT, int lane) {
  v16bf af[4];
#pragma unroll
  for (int kt = 0; kt < 4; ++kt) af[kt] = lda128(A, kt, lane);
  const int N = lane & 15, hi = lane >> 4;
#pragma unroll
  for (int nt = 0; nt < 8; ++nt) {
    v8f acc = zero8f();
#pragma unroll
    for (int kt = 0; kt < 4; ++kt)
      acc = wmma_bf16(af[kt], ldb_w(WT, nt, kt, lane), acc);
#pragma unroll
    for (int j = 0; j < 8; ++j)
      DT[(nt * 16 + N) * 16 + (j + 8 * hi)] = (bf16_t)acc[j];
  }
}

static __device__ __forceinline__ void mm_bias_relu(const bf16_t* A,
                                                    const bf16_t* WT,
                                                    const float* bias, bf16_t* D,
                                                    int lane) {
  v16bf af[4];
#pragma unroll
  for (int kt = 0; kt < 4; ++kt) af[kt] = lda128(A, kt, lane);
  const int N = lane & 15, hi = lane >> 4;
#pragma unroll
  for (int nt = 0; nt < 8; ++nt) {
    v8f acc = zero8f();
#pragma unroll
    for (int kt = 0; kt < 4; ++kt)
      acc = wmma_bf16(af[kt], ldb_w(WT, nt, kt, lane), acc);
    const float bb = bias[nt * 16 + N];
#pragma unroll
    for (int j = 0; j < 8; ++j)
      D[(j + 8 * hi) * 128 + nt * 16 + N] = (bf16_t)fmaxf(acc[j] + bb, 0.0f);
  }
}

// D = LayerNorm( RES + A@W (+ bias) ) * g + be   (stats over E=128 per row)
template <bool HAS_BIAS>
static __device__ __forceinline__ void mm_res_ln(const bf16_t* A, const bf16_t* WT,
                                                 const bf16_t* RES,
                                                 const float* bias, const float* g,
                                                 const float* be, bf16_t* D,
                                                 int lane) {
  v16bf af[4];
#pragma unroll
  for (int kt = 0; kt < 4; ++kt) af[kt] = lda128(A, kt, lane);
  const int N = lane & 15, hi = lane >> 4;
  float vals[8][8];
#pragma unroll
  for (int nt = 0; nt < 8; ++nt) {
    v8f acc = zero8f();
#pragma unroll
    for (int kt = 0; kt < 4; ++kt)
      acc = wmma_bf16(af[kt], ldb_w(WT, nt, kt, lane), acc);
    float bb = 0.0f;
    if (HAS_BIAS) bb = bias[nt * 16 + N];
#pragma unroll
    for (int j = 0; j < 8; ++j)
      vals[nt][j] = acc[j] + bb + (float)RES[(j + 8 * hi) * 128 + nt * 16 + N];
  }
  float mean[8], rstd[8];
#pragma unroll
  for (int j = 0; j < 8; ++j) {
    float s1 = 0.0f, s2 = 0.0f;
#pragma unroll
    for (int nt = 0; nt < 8; ++nt) {
      const float x = vals[nt][j];
      s1 += x;
      s2 += x * x;
    }
    s1 = rsum16(s1);
    s2 = rsum16(s2);
    const float mu = s1 * (1.0f / 128.0f);
    const float va = s2 * (1.0f / 128.0f) - mu * mu;
    mean[j] = mu;
    rstd[j] = rsqrtf(va + 1e-5f);
  }
#pragma unroll
  for (int nt = 0; nt < 8; ++nt) {
    const float gg = g[nt * 16 + N], bb = be[nt * 16 + N];
#pragma unroll
    for (int j = 0; j < 8; ++j)
      D[(j + 8 * hi) * 128 + nt * 16 + N] =
          (bf16_t)((vals[nt][j] - mean[j]) * rstd[j] * gg + bb);
  }
}

// ---------------------------------------------------------------------------
// Prologue: Wt[m][n][k] = bf16(W_m[k][n]) for the 8 weight matrices.
// ---------------------------------------------------------------------------
__global__ __launch_bounds__(256) void conflict_prep_weights(
    const float* __restrict__ Wq, const float* __restrict__ Wk,
    const float* __restrict__ Wv, const float* __restrict__ Wo,
    const float* __restrict__ W1, const float* __restrict__ W2,
    const float* __restrict__ Wqs, const float* __restrict__ Wks,
    bf16_t* __restrict__ WT) {
  const int idx = blockIdx.x * blockDim.x + threadIdx.x;  // 0 .. 131071
  const int m = idx >> 14;
  const int r = idx & 16383;
  const int n = r >> 7;
  const int k = r & 127;
  const float* s;
  switch (m) {
    case 0: s = Wq;  break;
    case 1: s = Wk;  break;
    case 2: s = Wv;  break;
    case 3: s = Wo;  break;
    case 4: s = W1;  break;
    case 5: s = W2;  break;
    case 6: s = Wqs; break;
    default: s = Wks; break;
  }
  WT[idx] = (bf16_t)s[k * 128 + n];
}

// ---------------------------------------------------------------------------
// Main kernel: 4 waves/block, one batch element per wave.
// ---------------------------------------------------------------------------
__global__ __launch_bounds__(128) void conflict_main(
    const float* __restrict__ agent, const float* __restrict__ city,
    const int* __restrict__ acts, const float* __restrict__ ln1g,
    const float* __restrict__ ln1b, const float* __restrict__ b1v,
    const float* __restrict__ b2v, const float* __restrict__ ln2g,
    const float* __restrict__ ln2b, const bf16_t* __restrict__ WT,
    float* __restrict__ out) {
  extern __shared__ char smem[];
  const int lane = threadIdx.x & 31;
  const int w = threadIdx.x >> 5;
  const int b = blockIdx.x * WAVES + w;

  char* ws = smem + (size_t)w * WAVE_BYTES;
  bf16_t* SEL = (bf16_t*)ws;        // 16x128 selected city rows
  bf16_t* AGL = SEL + 2048;         // 16x128 agent rows
  bf16_t* B0 = AGL + 2048;          // q -> h -> ks
  bf16_t* B1 = B0 + 2048;           // k -> ffn
  bf16_t* B2 = B1 + 2048;           // v^T (128x16) -> cac
  bf16_t* B3 = B2 + 2048;           // attn-out -> qs
  float* SCR = (float*)(B3 + 2048); // 256 f32 scratch (attn tile / out stage)
  int* LACTS = (int*)(SCR + 256);   // 16 ints

  // Pull the 256KB of bf16 weights toward the WGP (L2 resident anyway).
  for (int i = threadIdx.x; i < 2048; i += blockDim.x)
    __builtin_prefetch(WT + (size_t)i * 64, 0, 1);

  if (lane < 16) LACTS[lane] = acts[(size_t)b * 16 + lane];
  wavefence();

  // Per-lane column conflict mask: bit q of blockedbits == attention blocked
  // for (row q, col N). Identical in both 16-lane halves.
  unsigned blockedbits = 0;
  {
    const int n = lane & 15;
    const int an = LACTS[n];
#pragma unroll
    for (int q = 0; q < 16; ++q) {
      const int aq = LACTS[q];
      const bool conflict = (aq == 0) ? (q == n) : (aq == an);
      if (!conflict) blockedbits |= (1u << q);
    }
  }

  // Gather selected city rows + agent rows, convert f32 -> bf16 into LDS.
#pragma unroll
  for (int idx0 = 0; idx0 < 512; idx0 += 32) {
    const int idx = idx0 + lane;
    const int r = idx >> 5;
    const int c4 = (idx & 31) << 2;
    const float4 fa = *(const float4*)(agent + ((size_t)b * 16 + r) * 128 + c4);
    const int cr = LACTS[r];
    const float4 fs = *(const float4*)(city + ((size_t)b * 128 + cr) * 128 + c4);
    v4bf va, vs;
    va[0] = (bf16_t)fa.x; va[1] = (bf16_t)fa.y; va[2] = (bf16_t)fa.z; va[3] = (bf16_t)fa.w;
    vs[0] = (bf16_t)fs.x; vs[1] = (bf16_t)fs.y; vs[2] = (bf16_t)fs.z; vs[3] = (bf16_t)fs.w;
    *(v4bf*)(AGL + r * 128 + c4) = va;
    *(v4bf*)(SEL + r * 128 + c4) = vs;
  }
  wavefence();

  // S1-S3: q = SEL@Wq, k = AG@Wk, v^T = (AG@Wv)^T
  mm_store(SEL, WT + 0 * 16384, B0, lane);
  mm_store(AGL, WT + 1 * 16384, B1, lane);
  mm_storeT(AGL, WT + 2 * 16384, B2, lane);
  wavefence();

  // S4: 8-head attention, dh=16 (zero-padded K), softmax over 16 keys.
  const int N = lane & 15, hi = lane >> 4;
  bf16_t* ATT = (bf16_t*)SCR;  // 16x16 bf16 tile
#pragma unroll 1
  for (int h = 0; h < 8; ++h) {
    const v16bf aq = lda_k16(B0, 128, 16 * h, lane);
    const v16bf bk = ldb_rT_k16(B1, 128, 0, 16 * h, lane);
    v8f sc = wmma_bf16(aq, bk, zero8f());
    float p[8];
#pragma unroll
    for (int j = 0; j < 8; ++j) {
      const int Mr = j + 8 * hi;
      float x = sc[j] * 0.25f;  // 1/sqrt(dh)
      if ((blockedbits >> Mr) & 1u) x = NEGV;
      p[j] = x;
    }
#pragma unroll
    for (int j = 0; j < 8; ++j) {
      const float mx = rmax16(p[j]);
      const float e = __expf(p[j] - mx);
      const float s = rsum16(e);
      p[j] = e / s;
    }
    wavefence();
#pragma unroll
    for (int j = 0; j < 8; ++j) ATT[(j + 8 * hi) * 16 + N] = (bf16_t)p[j];
    wavefence();
    const v16bf aa = lda_k16(ATT, 16, 0, lane);
    const v16bf bv = ldb_rT_k16(B2, 16, 16 * h, 0, lane);
    v8f of = wmma_bf16(aa, bv, zero8f());
#pragma unroll
    for (int j = 0; j < 8; ++j)
      B3[(j + 8 * hi) * 128 + 16 * h + N] = (bf16_t)of[j];
  }
  wavefence();

  // S5: h = LN1(SEL + out@Wo)           -> B0
  mm_res_ln<false>(B3, WT + 3 * 16384, SEL, nullptr, ln1g, ln1b, B0, lane);
  wavefence();
  // S6: t = relu(h@W1 + b1)             -> B1
  mm_bias_relu(B0, WT + 4 * 16384, b1v, B1, lane);
  wavefence();
  // S7: cac = LN2(h + t@W2 + b2)        -> B2
  mm_res_ln<true>(B1, WT + 5 * 16384, B0, b2v, ln2g, ln2b, B2, lane);
  wavefence();
  // S8: qs = cac@Wq_s -> B3 ; S9: ks = AG@Wk_s -> B0
  mm_store(B2, WT + 6 * 16384, B3, lane);
  mm_store(AGL, WT + 7 * 16384, B0, lane);
  wavefence();

  // S10: logits = mask( 10*tanh( qs @ ks^T / sqrt(128) ) )
  {
    v16bf af[4];
#pragma unroll
    for (int kt = 0; kt < 4; ++kt) af[kt] = lda128(B3, kt, lane);
    v8f acc = zero8f();
#pragma unroll
    for (int kt = 0; kt < 4; ++kt)
      acc = wmma_bf16(af[kt], ldb_rT(B0, kt, lane), acc);
    const float s = 1.0f / sqrtf(128.0f);
#pragma unroll
    for (int j = 0; j < 8; ++j) {
      const int Mr = j + 8 * hi;
      float x = CLIPV * tanhf(acc[j] * s);
      if ((blockedbits >> Mr) & 1u) x = NEGV;
      SCR[Mr * 16 + N] = x;
    }
  }
  wavefence();
  // coalesced 32B stores of the 16x16 logits tile
  {
    float* dst = out + (size_t)b * 256 + lane * 8;
    const float4 a0 = *(const float4*)(SCR + lane * 8);
    const float4 a1 = *(const float4*)(SCR + lane * 8 + 4);
    *(float4*)dst = a0;
    *(float4*)(dst + 4) = a1;
  }
}

// ---------------------------------------------------------------------------
extern "C" void kernel_launch(void* const* d_in, const int* in_sizes, int n_in,
                              void* d_out, int out_size, void* d_ws,
                              size_t ws_size, hipStream_t stream) {
  (void)in_sizes; (void)n_in; (void)out_size; (void)ws_size;
  const float* agent = (const float*)d_in[0];
  const float* city = (const float*)d_in[1];
  const int* acts = (const int*)d_in[2];
  const float* Wq = (const float*)d_in[3];
  const float* Wk = (const float*)d_in[4];
  const float* Wv = (const float*)d_in[5];
  const float* Wo = (const float*)d_in[6];
  const float* ln1g = (const float*)d_in[7];
  const float* ln1b = (const float*)d_in[8];
  const float* W1 = (const float*)d_in[9];
  const float* b1 = (const float*)d_in[10];
  const float* W2 = (const float*)d_in[11];
  const float* b2 = (const float*)d_in[12];
  const float* ln2g = (const float*)d_in[13];
  const float* ln2b = (const float*)d_in[14];
  const float* Wqs = (const float*)d_in[15];
  const float* Wks = (const float*)d_in[16];
  bf16_t* WT = (bf16_t*)d_ws;  // 8 * 128*128 bf16 = 256 KB
  float* out = (float*)d_out;

  conflict_prep_weights<<<512, 256, 0, stream>>>(Wq, Wk, Wv, Wo, W1, W2, Wqs,
                                                 Wks, WT);
  conflict_main<<<4096 / WAVES, 32 * WAVES, WAVES * WAVE_BYTES, stream>>>(
      agent, city, acts, ln1g, ln1b, b1, b2, ln2g, ln2b, WT, out);
}
